// Compression_19043884990944
// MI455X (gfx1250) — matrix operand order repaired and checked by reading, not compiled, not run on hardware
//
#include <hip/hip_runtime.h>

typedef unsigned short u16;
typedef __attribute__((ext_vector_type(16))) __bf16 v16bf;
typedef __attribute__((ext_vector_type(8)))  float  v8f;
typedef __attribute__((ext_vector_type(4)))  int    v4i;

#define NROWS 8192
#define DDIM  1024
#define GAMMA 1e-4f
#define SCALE 0.03125f   /* 1/sqrt(1024) */

// LDS row strides (padded for bank-conflict-free strided ds_load_b128)
#define QI_STRIDE 1032   /* 1024 + 8  -> 2064B rows  */
#define QJ_STRIDE 264    /* 256 + 8   -> 528B rows   */
#define S_STRIDE  68     /* 64 + 4    -> 272B rows   */
#define P_STRIDE  72     /* 64 + 8    -> 144B rows   */

// ---- CDNA5 async global->LDS copy path (ASYNCcnt), with register fallback ----
#if defined(__has_builtin)
#if __has_builtin(__builtin_amdgcn_global_load_async_to_lds_b128) && \
    __has_builtin(__builtin_amdgcn_s_wait_asynccnt)
#define USE_ASYNC_LDS 1
#endif
#endif

#ifdef USE_ASYNC_LDS
typedef __attribute__((address_space(1))) v4i* gv4p;   // global int4*
typedef __attribute__((address_space(3))) v4i* lv4p;   // LDS int4*
static __device__ __forceinline__ void async_cp_b128(const u16* g, u16* l) {
    __builtin_amdgcn_global_load_async_to_lds_b128((gv4p)(u16*)g, (lv4p)l, 0, 0);
}
#define ASYNC_WAIT() __builtin_amdgcn_s_wait_asynccnt(0)
#else
#define ASYNC_WAIT() ((void)0)
#endif

static __device__ __forceinline__ u16 f2bf(float x) {
    unsigned v = __builtin_bit_cast(unsigned, x);
    unsigned r = v + 0x7FFFu + ((v >> 16) & 1u);   // round-to-nearest-even
    return (u16)(r >> 16);
}

union FragU {
    v16bf v;
    u16   u[16];
    uint4 q[2];
};

union V8U { v8f v; float f[8]; };

static __device__ __forceinline__ v16bf ldfrag(const u16* p0, const u16* p1) {
    FragU f;
    f.q[0] = *(const uint4*)p0;
    f.q[1] = *(const uint4*)p1;
    return f.v;
}

static __device__ __forceinline__ v8f wmma_bf16(v16bf a, v16bf b, v8f c) {
    return __builtin_amdgcn_wmma_f32_16x16x32_bf16(false, a, false, b, (short)0, c,
                                                   false, false);
}

static __device__ __forceinline__ v8f vzero8() {
    return (v8f){0.f,0.f,0.f,0.f,0.f,0.f,0.f,0.f};
}

// ---------------------------------------------------------------------------
// Kernel 1: X[N][D] fp32 -> Xb[N][D] bf16 (row major) + XbT[D][N] bf16
// ---------------------------------------------------------------------------
__global__ __launch_bounds__(256) void xpose_bf16(const float* __restrict__ X,
                                                  u16* __restrict__ Xb,
                                                  u16* __restrict__ XT) {
    __shared__ __align__(16) u16 tile[64][72];
    const int r0 = blockIdx.x * 64;   // over N
    const int c0 = blockIdx.y * 64;   // over D
    const int tx = threadIdx.x, ty = threadIdx.y;
#pragma unroll
    for (int i = 0; i < 4; ++i) {
        int r = ty * 4 + i;
        float4 f = *(const float4*)(X + (size_t)(r0 + r) * DDIM + c0 + tx * 4);
        ushort4 h;
        h.x = f2bf(f.x); h.y = f2bf(f.y); h.z = f2bf(f.z); h.w = f2bf(f.w);
        tile[r][tx*4+0] = h.x; tile[r][tx*4+1] = h.y;
        tile[r][tx*4+2] = h.z; tile[r][tx*4+3] = h.w;
        *(ushort4*)(Xb + (size_t)(r0 + r) * DDIM + c0 + tx * 4) = h;
    }
    __syncthreads();
#pragma unroll
    for (int i = 0; i < 4; ++i) {
        int d = ty * 4 + i;
        ushort4 h;
        h.x = tile[tx*4+0][d]; h.y = tile[tx*4+1][d];
        h.z = tile[tx*4+2][d]; h.w = tile[tx*4+3][d];
        *(ushort4*)(XT + (size_t)(c0 + d) * NROWS + r0 + tx * 4) = h;
    }
}

// ---------------------------------------------------------------------------
// Kernel 1b: W fp32 -> Wb bf16
// ---------------------------------------------------------------------------
__global__ __launch_bounds__(256) void wconv(const float* __restrict__ W,
                                             u16* __restrict__ Wb) {
    size_t i = ((size_t)blockIdx.x * 256 + threadIdx.x) * 4;
    float4 f = *(const float4*)(W + i);
    ushort4 h;
    h.x = f2bf(f.x); h.y = f2bf(f.y); h.z = f2bf(f.z); h.w = f2bf(f.w);
    *(ushort4*)(Wb + i) = h;
}

// ---------------------------------------------------------------------------
// Kernel 2: Q = X @ W^T (bf16 in -> bf16 out), pure-WMMA 128x128 block tile
// ---------------------------------------------------------------------------
__global__ __launch_bounds__(256) void qproj(const u16* __restrict__ Xb,
                                             const u16* __restrict__ Wb,
                                             u16* __restrict__ Qb) {
    const int lane = threadIdx.x & 31;
    const int wid  = threadIdx.x >> 5;
    const int m    = lane & 15;
    const int half = lane >> 4;
    const int wm = wid & 3, wn = wid >> 2;
    const int rbase = blockIdx.x * 128 + wm * 32;   // 2 M-tiles
    const int cbase = blockIdx.y * 128 + wn * 64;   // 4 N-tiles

    v8f acc[2][4];
#pragma unroll
    for (int i = 0; i < 2; ++i)
#pragma unroll
        for (int j = 0; j < 4; ++j) acc[i][j] = vzero8();

    for (int kk = 0; kk < DDIM; kk += 32) {
        v16bf a[2], b[4];
#pragma unroll
        for (int mt = 0; mt < 2; ++mt) {
            const u16* pr = Xb + (size_t)(rbase + mt * 16 + m) * DDIM + kk;
            a[mt] = ldfrag(pr + half * 8, pr + 16 + half * 8);
        }
#pragma unroll
        for (int nt = 0; nt < 4; ++nt) {
            const u16* pw = Wb + (size_t)(cbase + nt * 16 + m) * DDIM + kk;
            b[nt] = ldfrag(pw + half * 8, pw + 16 + half * 8);
        }
#pragma unroll
        for (int mt = 0; mt < 2; ++mt)
#pragma unroll
            for (int nt = 0; nt < 4; ++nt)
                acc[mt][nt] = wmma_bf16(a[mt], b[nt], acc[mt][nt]);
    }
#pragma unroll
    for (int mt = 0; mt < 2; ++mt)
#pragma unroll
        for (int nt = 0; nt < 4; ++nt) {
            V8U c; c.v = acc[mt][nt];
#pragma unroll
            for (int r = 0; r < 8; ++r) {
                int row = rbase + mt * 16 + r + 8 * half;
                int col = cbase + nt * 16 + m;
                Qb[(size_t)row * DDIM + col] = f2bf(c.f[r]);
            }
        }
}

// ---------------------------------------------------------------------------
// Kernel 3: out = X + GAMMA * softmax(Q Q^T / 32) @ X   (flash streaming)
// 64 rows/block; Qi resident in LDS; Qj K-slices double-buffered in LDS
// via async global->LDS copies (ASYNCcnt) when available.
// ---------------------------------------------------------------------------
__global__ __launch_bounds__(256) void attn(const float* __restrict__ X,
                                            const u16* __restrict__ Qb,
                                            const u16* __restrict__ XbT,
                                            float* __restrict__ out) {
    __shared__ __align__(16) u16   QiL[64 * QI_STRIDE];      // 129 KB resident Q_i
    __shared__ __align__(16) u16   QjL[2][64 * QJ_STRIDE];   // 66 KB staged Q_j
    __shared__ __align__(16) float Sld[64 * S_STRIDE];       // 17 KB scores
    __shared__ __align__(16) u16   Pld[64 * P_STRIDE];       // 9 KB probs (bf16)
    __shared__ float pm[256], ps_[256];
    __shared__ float mrow[64], lrow[64], arow[64], mnrow[64];

    const int tid  = threadIdx.x;
    const int lane = tid & 31;
    const int wid  = tid >> 5;
    const int m    = lane & 15;
    const int half = lane >> 4;
    const int rblk = blockIdx.x * 64;

    const int swm = wid & 3;      // S-phase: 4x2 wave grid over 64x64 chunk
    const int swn = wid >> 2;
    const int dbase = wid * 128;  // PV-phase: 128-wide D slice per wave

    const int srow = tid >> 2, sseg = tid & 3;   // softmax + staging coords

    // ---- load resident Q_i block into LDS ----
    {
        const u16* src = Qb + (size_t)(rblk + srow) * DDIM + sseg * 256;
        u16* dst = QiL + srow * QI_STRIDE + sseg * 256;
#ifdef USE_ASYNC_LDS
#pragma unroll
        for (int i = 0; i < 32; ++i) async_cp_b128(src + i * 8, dst + i * 8);
#else
#pragma unroll
        for (int i = 0; i < 32; ++i) ((uint4*)dst)[i] = ((const uint4*)src)[i];
#endif
    }
    v8f O[4][8];
#pragma unroll
    for (int i = 0; i < 4; ++i)
#pragma unroll
        for (int j = 0; j < 8; ++j) O[i][j] = vzero8();
    if (tid < 64) { mrow[tid] = -1e30f; lrow[tid] = 0.f; }

    for (int jb = 0; jb < NROWS; jb += 64) {
        // ---- stage 0 of Q_j slice ----
        {
            const u16* src = Qb + (size_t)(jb + srow) * DDIM + sseg * 64;
            u16* dst = QjL[0] + srow * QJ_STRIDE + sseg * 64;
#ifdef USE_ASYNC_LDS
#pragma unroll
            for (int i = 0; i < 8; ++i) async_cp_b128(src + i * 8, dst + i * 8);
#else
#pragma unroll
            for (int i = 0; i < 8; ++i) ((uint4*)dst)[i] = ((const uint4*)src)[i];
#endif
        }
        ASYNC_WAIT();
        __syncthreads();

        // ---- S = Q_i @ Q_j^T, K staged in 4 slices of 256 ----
        v8f s0 = vzero8(), s1 = vzero8();
        for (int st = 0; st < 4; ++st) {
            if (st < 3) {
                const u16* src = Qb + (size_t)(jb + srow) * DDIM
                               + (st + 1) * 256 + sseg * 64;
                u16* dst = QjL[(st + 1) & 1] + srow * QJ_STRIDE + sseg * 64;
#ifdef USE_ASYNC_LDS
#pragma unroll
                for (int i = 0; i < 8; ++i) async_cp_b128(src + i * 8, dst + i * 8);
#else
#pragma unroll
                for (int i = 0; i < 8; ++i) ((uint4*)dst)[i] = ((const uint4*)src)[i];
#endif
            }
            const u16* qi = QiL + (swm * 16 + m) * QI_STRIDE + st * 256;
            const u16* qj = QjL[st & 1] + (swn * 32 + m) * QJ_STRIDE;
#pragma unroll
            for (int k2 = 0; k2 < 256; k2 += 32) {
                v16bf a = ldfrag(qi + k2 + half * 8, qi + k2 + 16 + half * 8);
                const u16* pb0 = qj + k2 + half * 8;
                s0 = wmma_bf16(a, ldfrag(pb0, pb0 + 16), s0);
                const u16* pb1 = pb0 + 16 * QJ_STRIDE;
                s1 = wmma_bf16(a, ldfrag(pb1, pb1 + 16), s1);
            }
            ASYNC_WAIT();
            __syncthreads();
        }
        // ---- scatter S to LDS ----
        {
            V8U c0, c1; c0.v = s0; c1.v = s1;
#pragma unroll
            for (int r = 0; r < 8; ++r) {
                int row = swm * 16 + r + 8 * half;
                Sld[row * S_STRIDE + swn * 32 + m]      = c0.f[r];
                Sld[row * S_STRIDE + swn * 32 + 16 + m] = c1.f[r];
            }
        }
        __syncthreads();

        // ---- online softmax: 4 threads per row ----
        float sreg[16];
        float lm = -1e30f;
#pragma unroll
        for (int i = 0; i < 16; ++i) {
            sreg[i] = Sld[srow * S_STRIDE + sseg * 16 + i];
            lm = fmaxf(lm, sreg[i]);
        }
        pm[tid] = lm;
        __syncthreads();
        {
            float cm = fmaxf(fmaxf(pm[srow*4], pm[srow*4+1]),
                             fmaxf(pm[srow*4+2], pm[srow*4+3]));
            float mo = mrow[srow];
            float mn = fmaxf(mo, cm * SCALE);
            float al = __expf(mo - mn);
            float ssum = 0.f;
#pragma unroll
            for (int i = 0; i < 16; ++i) {
                float p = __expf(sreg[i] * SCALE - mn);
                ssum += p;
                Pld[srow * P_STRIDE + sseg * 16 + i] = f2bf(p);
            }
            ps_[tid] = ssum;
            if (sseg == 0) { arow[srow] = al; mnrow[srow] = mn; }
        }
        __syncthreads();
        if (tid < 64) {
            lrow[tid] = lrow[tid] * arow[tid]
                      + ps_[tid*4] + ps_[tid*4+1] + ps_[tid*4+2] + ps_[tid*4+3];
            mrow[tid] = mnrow[tid];
        }

        // ---- rescale O by alpha ----
#pragma unroll
        for (int mt = 0; mt < 4; ++mt) {
            V8U c0[8];
#pragma unroll
            for (int nt = 0; nt < 8; ++nt) c0[nt].v = O[mt][nt];
#pragma unroll
            for (int r = 0; r < 8; ++r) {
                float al = arow[mt * 16 + r + 8 * half];
#pragma unroll
                for (int nt = 0; nt < 8; ++nt) c0[nt].f[r] *= al;
            }
#pragma unroll
            for (int nt = 0; nt < 8; ++nt) O[mt][nt] = c0[nt].v;
        }
        // ---- O += P @ X_j ----
#pragma unroll
        for (int kk = 0; kk < 64; kk += 32) {
            v16bf b[8];
#pragma unroll
            for (int nt = 0; nt < 8; ++nt) {
                const u16* pb = XbT + (size_t)(dbase + nt * 16 + m) * NROWS
                                    + jb + kk + half * 8;
                b[nt] = ldfrag(pb, pb + 16);
            }
#pragma unroll
            for (int mt = 0; mt < 4; ++mt) {
                const u16* pa = Pld + (mt * 16 + m) * P_STRIDE + kk;
                v16bf a = ldfrag(pa + half * 8, pa + 16 + half * 8);
#pragma unroll
                for (int nt = 0; nt < 8; ++nt)
                    O[mt][nt] = wmma_bf16(a, b[nt], O[mt][nt]);
            }
        }
    }

    // ---- epilogue: out = X + GAMMA * O / l ----
#pragma unroll
    for (int mt = 0; mt < 4; ++mt)
#pragma unroll
        for (int nt = 0; nt < 8; ++nt) {
            V8U c; c.v = O[mt][nt];
#pragma unroll
            for (int r = 0; r < 8; ++r) {
                int row = rblk + mt * 16 + r + 8 * half;
                float inv = 1.0f / lrow[mt * 16 + r + 8 * half];
                int col = dbase + nt * 16 + m;
                size_t idx = (size_t)row * DDIM + col;
                out[idx] = X[idx] + GAMMA * c.f[r] * inv;
            }
        }
}

// ---------------------------------------------------------------------------
extern "C" void kernel_launch(void* const* d_in, const int* in_sizes, int n_in,
                              void* d_out, int out_size, void* d_ws, size_t ws_size,
                              hipStream_t stream) {
    (void)in_sizes; (void)n_in; (void)out_size; (void)ws_size;
    const float* X = (const float*)d_in[0];
    const float* W = (const float*)d_in[1];
    float* out = (float*)d_out;

    char* ws = (char*)d_ws;
    u16* XbT = (u16*)ws;                                  // [D][N] bf16, 16 MB
    u16* Qb  = (u16*)(ws + (size_t)16 * 1024 * 1024);     // [N][D] bf16, 16 MB
    u16* Xb  = (u16*)(ws + (size_t)32 * 1024 * 1024);     // [N][D] bf16, 16 MB
    u16* Wb  = (u16*)(ws + (size_t)48 * 1024 * 1024);     // [D][D] bf16,  2 MB

    xpose_bf16<<<dim3(NROWS / 64, DDIM / 64), dim3(16, 16), 0, stream>>>(X, Xb, XbT);
    wconv<<<(DDIM * DDIM) / (256 * 4), 256, 0, stream>>>(W, Wb);
    qproj<<<dim3(NROWS / 128, DDIM / 128), 256, 0, stream>>>(Xb, Wb, Qb);
    attn<<<NROWS / 64, 256, 0, stream>>>(X, Qb, XbT, out);
}
